// Convolution_56908316672259
// MI455X (gfx1250) — matrix operand order repaired and checked by reading, not compiled, not run on hardware
//
#include <hip/hip_runtime.h>

typedef __attribute__((ext_vector_type(16))) _Float16 v16h;
typedef __attribute__((ext_vector_type(8)))  _Float16 v8h;
typedef __attribute__((ext_vector_type(8)))  float    v8f;

#define INV_SQRT3 0.57735026918962576f
#define S_IN      0.17677669529663687f   // 1/sqrt(32)
#define S_MID     0.125f                 // 1/sqrt(64)
#define S_SE      0.088388347648318433f  // 1/sqrt(128)
#define INV_NN    0.35355339059327373f   // 1/sqrt(8)
#define BWAVES    2

// wave-local LDS ordering: same wave wrote, same wave reads (cross-lane).
__device__ __forceinline__ void wave_lds_fence() {
    __builtin_amdgcn_wave_barrier();
    asm volatile("s_wait_dscnt 0" ::: "memory");
    __builtin_amdgcn_wave_barrier();
}

__device__ __forceinline__ v16h cat8(v8h a, v8h b) {
    return __builtin_shufflevector(a, b, 0,1,2,3,4,5,6,7,8,9,10,11,12,13,14,15);
}

// ---------------- Kernel A: node-side linears (node_self -> d_out, f0/f1 -> ws)
__global__ __launch_bounds__(256) void node_linear_kernel(
    const float* __restrict__ node_input, const float* __restrict__ node_attr,
    const float* __restrict__ w_sc0, const float* __restrict__ w_sc1,
    const float* __restrict__ w_lin1_0, const float* __restrict__ w_lin1_1,
    float* __restrict__ out_node, float* __restrict__ f0, float* __restrict__ f1, int N)
{
    __shared__ float s_sc0[1024], s_sc1[1024], s_l0[1024], s_l1[1024];
    int tid = threadIdx.x;
    for (int i = tid; i < 1024; i += blockDim.x) {
        s_sc0[i] = w_sc0[i]; s_sc1[i] = w_sc1[i];
        s_l0[i]  = w_lin1_0[i]; s_l1[i] = w_lin1_1[i];
    }
    __syncthreads();
    int gid = blockIdx.x * blockDim.x + tid;
    int n = gid >> 5, w = gid & 31;
    if (n >= N) return;
    const float* row = node_input + (size_t)n * 128;
    float a0 = 0.f, l0 = 0.f;
    float a1[3] = {0.f,0.f,0.f}, l1[3] = {0.f,0.f,0.f};
    for (int u = 0; u < 32; ++u) {
        float x0 = row[u];
        a0 = fmaf(x0, s_sc0[u*32+w], a0);
        l0 = fmaf(x0, s_l0[u*32+w],  l0);
        float wsc1 = s_sc1[u*32+w], wl1 = s_l1[u*32+w];
#pragma unroll
        for (int m = 0; m < 3; ++m) {
            float x1 = row[32 + u*3 + m];
            a1[m] = fmaf(x1, wsc1, a1[m]);
            l1[m] = fmaf(x1, wl1,  l1[m]);
        }
    }
    float a = node_attr[n] * S_IN;
    out_node[(size_t)n*128 + w] = a0 * a;
    f0[(size_t)n*32 + w] = l0 * a;
#pragma unroll
    for (int m = 0; m < 3; ++m) {
        out_node[(size_t)n*128 + 32 + w*3 + m] = a1[m] * a;
        f1[((size_t)n*32 + w)*3 + m] = l1[m] * a;
    }
}

// ---------------- Kernel Z: zero the n0/n1 accumulators
__global__ void zero_kernel(float* __restrict__ p, size_t count) {
    size_t i = (size_t)blockIdx.x * blockDim.x + threadIdx.x;
    size_t stride = (size_t)gridDim.x * blockDim.x;
    for (; i < count; i += stride) p[i] = 0.f;
}

// ---------------- Kernel B: edge FC via WMMA + edgewise tensor products + scatter
__global__ __launch_bounds__(64) void edge_kernel(
    const int* __restrict__ edge_src, const int* __restrict__ edge_dst,
    const float* __restrict__ edge_attr, const float* __restrict__ edge_scalars,
    const float* __restrict__ fc_w1, const float* __restrict__ fc_w2,
    const float* __restrict__ w_se0a, const float* __restrict__ w_se0b,
    const float* __restrict__ w_se1a, const float* __restrict__ w_se1b,
    const float* __restrict__ f0, const float* __restrict__ f1,
    float* __restrict__ n0, float* __restrict__ n1,
    float* __restrict__ edge_out, int E)
{
    // fc_w1 transposed + K-padded: [n=64][k=32], zeros for k>=16
    __shared__ __align__(16) _Float16 s_w1t[64*32];
    // fc_w2 transposed: [n=128][k=64]
    __shared__ __align__(16) _Float16 s_w2t[128*64];
    __shared__ float s_sea0[64], s_seb0[64], s_sea1[64], s_seb1[64];
    // per-wave edge_scalars tile as f16, K-padded: [M=16][k=32], zeros for k>=16
    __shared__ __align__(16) _Float16 s_es[BWAVES][16*32];
    __shared__ __align__(16) _Float16 s_h [BWAVES][16*64];  // hidden layer (f16)
    __shared__ __align__(16) float    s_w [BWAVES][16*128]; // FC output tile

    int tid = threadIdx.x;
    // stage fc_w1 transposed+padded (strided global reads; one-time, L2-resident)
    for (int i = tid; i < 64*32; i += blockDim.x) {
        int n = i >> 5, k = i & 31;
        s_w1t[i] = (k < 16) ? (_Float16)fc_w1[k*64 + n] : (_Float16)0.f;
    }
    // stage fc_w2 transposed: coalesced global read, strided LDS write
    for (int i = tid; i < 64*128; i += blockDim.x) {
        int k = i >> 7, n = i & 127;
        s_w2t[n*64 + k] = (_Float16)fc_w2[i];
    }
    for (int i = tid; i < 64; i += blockDim.x) {
        s_sea0[i] = w_se0a[i]; s_seb0[i] = w_se0b[i];
        s_sea1[i] = w_se1a[i]; s_seb1[i] = w_se1b[i];
    }
    int lane = tid & 31, wv = tid >> 5;
    // zero the K-pad region of the per-wave es tile once (persists across tiles)
    {
        _Float16* esw = s_es[wv];
#pragma unroll
        for (int j = 0; j < 16; ++j) esw[lane*16 + j] = (_Float16)0.f;
    }
    __syncthreads();

    int waveGlobal = blockIdx.x * BWAVES + wv;
    int waveCount  = gridDim.x * BWAVES;
    int nTiles = (E + 15) >> 4;
    _Float16* es   = s_es[wv];
    _Float16* hbuf = s_h[wv];
    float*    wbuf = s_w[wv];
    int M  = lane & 15;      // row within tile / edge slot
    int hi = lane >> 4;      // half-wave id

    for (int tile = waveGlobal; tile < nTiles; tile += waveCount) {
        int ebase = tile * 16;
        // stage edge_scalars tile: lane loads 8 contiguous floats -> f16
        {
            int e = ebase + (lane >> 1);
            int off = (lane & 1) * 8;
#pragma unroll
            for (int j = 0; j < 8; ++j)
                es[(lane >> 1)*32 + off + j] =
                    (e < E) ? (_Float16)edge_scalars[(size_t)e*16 + off + j]
                            : (_Float16)0.f;
        }
        wave_lds_fence();

        // A operand (16x32 f16): lane M=lane%16 holds K = 8*hi+0..7 and 16+8*hi+0..7
        v16h a1 = cat8(*(const v8h*)(es + M*32 + 8*hi),
                       *(const v8h*)(es + M*32 + 16 + 8*hi));
        // GEMM1: h = sin(0.25 * es @ fc_w1)   (4 N-tiles of 16)
#pragma unroll
        for (int t1 = 0; t1 < 4; ++t1) {
            // B operand: lane holds col n = t1*16+M, K = 8*hi+0..7 and 16+8*hi+0..7
            const _Float16* bp = s_w1t + (t1*16 + M)*32 + 8*hi;
            v16h b = cat8(*(const v8h*)bp, *(const v8h*)(bp + 16));
            v8f c = {};
            c = __builtin_amdgcn_wmma_f32_16x16x32_f16(false, a1, false, b,
                                                       (short)0, c, false, false);
#pragma unroll
            for (int r = 0; r < 8; ++r)
                hbuf[((hi << 3) + r)*64 + t1*16 + M] =
                    (_Float16)__sinf(c[r] * 0.25f);   // hw v_sin_f32: args are O(1)
        }
        wave_lds_fence();

        // A operands for GEMM2 (K=0..31 and K=32..63 of h), already f16 in LDS
        v16h a2a = cat8(*(const v8h*)(hbuf + M*64 + 8*hi),
                        *(const v8h*)(hbuf + M*64 + 16 + 8*hi));
        v16h a2b = cat8(*(const v8h*)(hbuf + M*64 + 32 + 8*hi),
                        *(const v8h*)(hbuf + M*64 + 48 + 8*hi));
        // GEMM2: w = 0.125 * h @ fc_w2   (8 N-tiles x 2 K-chunks)
#pragma unroll
        for (int t2 = 0; t2 < 8; ++t2) {
            const _Float16* bp = s_w2t + (t2*16 + M)*64 + 8*hi;
            v16h ba = cat8(*(const v8h*)bp,        *(const v8h*)(bp + 16));
            v16h bb = cat8(*(const v8h*)(bp + 32), *(const v8h*)(bp + 48));
            v8f c = {};
            c = __builtin_amdgcn_wmma_f32_16x16x32_f16(false, a2a, false, ba,
                                                       (short)0, c, false, false);
            c = __builtin_amdgcn_wmma_f32_16x16x32_f16(false, a2b, false, bb,
                                                       (short)0, c, false, false);
#pragma unroll
            for (int r = 0; r < 8; ++r)
                wbuf[((hi << 3) + r)*128 + t2*16 + M] = c[r] * 0.125f;
        }
        wave_lds_fence();

        // ---- edgewise: lane -> edge M, channel half hi (u in [hi*16, hi*16+16))
        int e = ebase + M;
        bool valid = (e < E);
        int esrc = valid ? edge_src[e] : 0;
        int edst = valid ? edge_dst[e] : 0;
        float ea0  = valid ? edge_attr[(size_t)e*4 + 0] : 0.f;
        float ea1x = valid ? edge_attr[(size_t)e*4 + 1] : 0.f;
        float ea1y = valid ? edge_attr[(size_t)e*4 + 2] : 0.f;
        float ea1z = valid ? edge_attr[(size_t)e*4 + 3] : 0.f;
        const float* wrow  = wbuf + M*128;
        const float* g0row = f0 + (size_t)esrc*32;
        const float* g1row = f1 + (size_t)esrc*96;
        float* p0 = n0 + (size_t)edst*64;
        float* p1 = n1 + (size_t)edst*192;
        float se0a = 0.f, se0b = 0.f, se1a = 0.f;
        float se1b0 = 0.f, se1b1 = 0.f, se1b2 = 0.f;
        for (int uu = 0; uu < 16; ++uu) {
            int u = (hi << 4) + uu;
            float wA = wrow[u], wB = wrow[32+u], wC = wrow[64+u], wD = wrow[96+u];
            float g0v = g0row[u];
            float g1x = g1row[u*3+0], g1y = g1row[u*3+1], g1z = g1row[u*3+2];
            float mA = wA * g0v * ea0;
            float dot = g1x*ea1x + g1y*ea1y + g1z*ea1z;
            float mD = wD * dot * INV_SQRT3;
            float wBg = wB * g0v;
            float mBx = wBg*ea1x, mBy = wBg*ea1y, mBz = wBg*ea1z;
            float wCe = wC * ea0;
            float mCx = wCe*g1x, mCy = wCe*g1y, mCz = wCe*g1z;
            if (valid) {
                atomicAdd(p0 + u,        mA);
                atomicAdd(p0 + 32 + u,   mD);
                atomicAdd(p1 + u*3+0, mBx); atomicAdd(p1 + u*3+1, mBy); atomicAdd(p1 + u*3+2, mBz);
                atomicAdd(p1 + (32+u)*3+0, mCx); atomicAdd(p1 + (32+u)*3+1, mCy); atomicAdd(p1 + (32+u)*3+2, mCz);
            }
            se0a += mA*s_sea0[u] + mD*s_sea0[32+u];
            se0b += (mBx*ea1x + mBy*ea1y + mBz*ea1z)*s_seb0[u]
                  + (mCx*ea1x + mCy*ea1y + mCz*ea1z)*s_seb0[32+u];
            se1a += mA*s_sea1[u] + mD*s_sea1[32+u];
            se1b0 += mBx*s_seb1[u] + mCx*s_seb1[32+u];
            se1b1 += mBy*s_seb1[u] + mCy*s_seb1[32+u];
            se1b2 += mBz*s_seb1[u] + mCz*s_seb1[32+u];
        }
        se0a  += __shfl_xor(se0a, 16, 32);
        se0b  += __shfl_xor(se0b, 16, 32);
        se1a  += __shfl_xor(se1a, 16, 32);
        se1b0 += __shfl_xor(se1b0, 16, 32);
        se1b1 += __shfl_xor(se1b1, 16, 32);
        se1b2 += __shfl_xor(se1b2, 16, 32);
        if (hi == 0 && valid) {
            float se0 = (se0a*ea0  + se0b *INV_SQRT3) * S_SE;
            float s1x = (se1a*ea1x + se1b0*ea0) * S_SE;
            float s1y = (se1a*ea1y + se1b1*ea0) * S_SE;
            float s1z = (se1a*ea1z + se1b2*ea0) * S_SE;
            edge_out[(size_t)e*4 + 0] = ea0  + se0*INV_NN;
            edge_out[(size_t)e*4 + 1] = ea1x + s1x*INV_NN;
            edge_out[(size_t)e*4 + 2] = ea1y + s1y*INV_NN;
            edge_out[(size_t)e*4 + 3] = ea1z + s1z*INV_NN;
        }
    }
}

// ---------------- Kernel C: node output linears + alpha gating (RMW on d_out)
__global__ __launch_bounds__(256) void node_out_kernel(
    const float* __restrict__ node_attr,
    const float* __restrict__ w_lin2_0, const float* __restrict__ w_lin2_1,
    const float* __restrict__ w_alpha,
    const float* __restrict__ n0, const float* __restrict__ n1,
    float* __restrict__ out_node, int N)
{
    __shared__ float s_l0[2048], s_l1[2048], s_al[64];
    int tid = threadIdx.x;
    for (int i = tid; i < 2048; i += blockDim.x) { s_l0[i] = w_lin2_0[i]; s_l1[i] = w_lin2_1[i]; }
    if (tid < 64) s_al[tid] = w_alpha[tid];
    __syncthreads();
    int gid = blockIdx.x * blockDim.x + tid;
    int n = gid >> 5, w = gid & 31;
    if (n >= N) return;
    const float* r0 = n0 + (size_t)n*64;
    const float* r1 = n1 + (size_t)n*192;
    float acc0 = 0.f, accA = 0.f, acc1[3] = {0.f,0.f,0.f};
    for (int u = 0; u < 64; ++u) {
        float v0 = r0[u];
        acc0 = fmaf(v0, s_l0[u*32+w], acc0);
        accA = fmaf(v0, s_al[u], accA);
        float wl = s_l1[u*32+w];
        acc1[0] = fmaf(r1[u*3+0], wl, acc1[0]);
        acc1[1] = fmaf(r1[u*3+1], wl, acc1[1]);
        acc1[2] = fmaf(r1[u*3+2], wl, acc1[2]);
    }
    float scale = node_attr[n] * S_MID * INV_NN;  // fold segment-sum inv_nn here
    float out0  = acc0 * scale;
    float alpha = accA * scale;
    float* orow = out_node + (size_t)n*128;
    orow[w] += alpha * out0;
#pragma unroll
    for (int m = 0; m < 3; ++m)
        orow[32 + w*3 + m] += alpha * acc1[m] * scale;
}

extern "C" void kernel_launch(void* const* d_in, const int* in_sizes, int n_in,
                              void* d_out, int out_size, void* d_ws, size_t ws_size,
                              hipStream_t stream) {
    const float* node_input   = (const float*)d_in[0];
    const float* node_attr    = (const float*)d_in[1];
    const int*   edge_src     = (const int*)  d_in[2];
    const int*   edge_dst     = (const int*)  d_in[3];
    const float* edge_attr    = (const float*)d_in[4];
    const float* edge_scalars = (const float*)d_in[5];
    const float* w_sc0   = (const float*)d_in[6];
    const float* w_sc1   = (const float*)d_in[7];
    const float* w_lin10 = (const float*)d_in[8];
    const float* w_lin11 = (const float*)d_in[9];
    const float* fc_w1   = (const float*)d_in[10];
    const float* fc_w2   = (const float*)d_in[11];
    const float* w_lin20 = (const float*)d_in[12];
    const float* w_lin21 = (const float*)d_in[13];
    const float* w_alpha = (const float*)d_in[14];
    const float* w_se0a  = (const float*)d_in[15];
    const float* w_se0b  = (const float*)d_in[16];
    const float* w_se1a  = (const float*)d_in[17];
    const float* w_se1b  = (const float*)d_in[18];

    int N = in_sizes[0] / 128;
    int E = in_sizes[2];

    float* f0 = (float*)d_ws;                    // N*32
    float* f1 = f0 + (size_t)N*32;               // N*96
    float* n0 = f1 + (size_t)N*96;               // N*64
    float* n1 = n0 + (size_t)N*64;               // N*192
    float* out_node = (float*)d_out;             // N*128
    float* out_edge = out_node + (size_t)N*128;  // E*4

    zero_kernel<<<2048, 256, 0, stream>>>(n0, (size_t)N*256);
    node_linear_kernel<<<(N*32 + 255)/256, 256, 0, stream>>>(
        node_input, node_attr, w_sc0, w_sc1, w_lin10, w_lin11, out_node, f0, f1, N);
    edge_kernel<<<1024, 64, 0, stream>>>(
        edge_src, edge_dst, edge_attr, edge_scalars, fc_w1, fc_w2,
        w_se0a, w_se0b, w_se1a, w_se1b, f0, f1, n0, n1, out_edge, E);
    node_out_kernel<<<(N*32 + 255)/256, 256, 0, stream>>>(
        node_attr, w_lin20, w_lin21, w_alpha, n0, n1, out_node, N);
}
